// CPG_encoder_56753697849883
// MI455X (gfx1250) — matrix operand order repaired and checked by reading, not compiled, not run on hardware
//
#include <hip/hip_runtime.h>

#define BB 2
#define NN 10000
#define CC 128
#define KNB 8
#define TS 1024

typedef __attribute__((ext_vector_type(16))) __bf16 v16bf;
typedef __attribute__((ext_vector_type(8)))  float  v8f;
typedef __attribute__((ext_vector_type(4)))  float  v4f;

union AFrag { v4f q[2]; v16bf v; };

// ---------------------------------------------------------------------------
// Kernel 0: transpose + convert weights to bf16 (W^T[c][k]) for WMMA B frags.
// ---------------------------------------------------------------------------
__global__ void prep_wt_kernel(const float* __restrict__ Wfeat,
                               const float* __restrict__ Wft,
                               __bf16* __restrict__ WTfeat,
                               __bf16* __restrict__ WTft) {
  int idx = blockIdx.x * blockDim.x + threadIdx.x;
  if (idx < CC * CC) {
    int k = idx >> 7, c = idx & 127;
    WTfeat[c * CC + k] = (__bf16)Wfeat[idx];
    WTft[c * CC + k]   = (__bf16)Wft[idx];
  }
}

// ---------------------------------------------------------------------------
// Kernel 1: KNN (top-8 smallest squared distance, stable ties) per scene.
// LDS-tiled candidates; register-resident sorted top-8 (unrolled swap net).
// ---------------------------------------------------------------------------
__global__ __launch_bounds__(256) void knn_kernel(const float* __restrict__ coords,
                                                  int* __restrict__ knn) {
  __shared__ float lx[TS], ly[TS], lz[TS];
  const int scene = blockIdx.y;
  const int q = blockIdx.x * blockDim.x + threadIdx.x;
  const bool active = q < NN;
  float qx = 0.f, qy = 0.f, qz = 0.f;
  if (active) {
    int b = (scene * NN + q) * 4;
    qx = coords[b + 1]; qy = coords[b + 2]; qz = coords[b + 3];
  }
  float bd[KNB]; int bi[KNB];
#pragma unroll
  for (int k = 0; k < KNB; ++k) { bd[k] = 3.0e38f; bi[k] = 0; }

  for (int t0 = 0; t0 < NN; t0 += TS) {
    int lim = NN - t0; if (lim > TS) lim = TS;
    __syncthreads();
    for (int i = threadIdx.x; i < lim; i += blockDim.x) {
      int b = (scene * NN + t0 + i) * 4;
      lx[i] = coords[b + 1]; ly[i] = coords[b + 2]; lz[i] = coords[b + 3];
    }
    __syncthreads();
    if (active) {
      for (int i = 0; i < lim; ++i) {
        float dx = lx[i] - qx, dy = ly[i] - qy, dz = lz[i] - qz;
        float d = fmaf(dx, dx, fmaf(dy, dy, dz * dz));
        if (d < bd[KNB - 1]) {
          bd[KNB - 1] = d; bi[KNB - 1] = t0 + i;
#pragma unroll
          for (int j = KNB - 1; j > 0; --j) {
            if (bd[j] < bd[j - 1]) {
              float td = bd[j]; bd[j] = bd[j - 1]; bd[j - 1] = td;
              int   ti = bi[j]; bi[j] = bi[j - 1]; bi[j - 1] = ti;
            }
          }
        }
      }
    }
  }
  if (active) {
#pragma unroll
    for (int k = 0; k < KNB; ++k)
      knn[(scene * NN + q) * KNB + k] = scene * NN + bi[k];  // global row idx
  }
}

// ---------------------------------------------------------------------------
// Kernel 2: fused gather + 2x WMMA GEMM + softmax(K) + weighted sum +
//           residual + LayerNorm. 1 workgroup = 16 points, 8 waves.
// ---------------------------------------------------------------------------
__device__ __forceinline__ void gemm_tile(const __bf16* __restrict__ arow,
                                          const __bf16* __restrict__ WT,
                                          int nloc, int h, v8f acc[8]) {
#pragma unroll
  for (int kk = 0; kk < 4; ++kk) {
    AFrag af;
    af.q[0] = *(const v4f*)(arow + kk * 32 + h * 16);
    af.q[1] = *(const v4f*)(arow + kk * 32 + h * 16 + 8);
#pragma unroll
    for (int j = 0; j < 8; ++j) {
      v16bf bfr = *(const v16bf*)(WT + (16 * j + nloc) * CC + kk * 32 + h * 16);
      acc[j] = __builtin_amdgcn_wmma_f32_16x16x32_bf16(
          false, af.v, false, bfr, (short)0, acc[j], false, false);
    }
  }
}

__global__ __launch_bounds__(256) void attn_kernel(
    const float* __restrict__ F, const float* __restrict__ coords,
    const int* __restrict__ knn,
    const __bf16* __restrict__ WTfeat, const __bf16* __restrict__ WTft,
    const float* __restrict__ Wc, const float* __restrict__ b_coord,
    const float* __restrict__ b_feat, const float* __restrict__ b_ft,
    const float* __restrict__ gamma_, const float* __restrict__ beta_,
    float* __restrict__ out) {
  __shared__ __align__(32) __bf16 As[CC * CC];  // 128 rows x 128 K (bf16) = 32KB
  __shared__ float relc[128 * 4];               // per-row relative coords

  const int tid = threadIdx.x;
  const int scene = blockIdx.y;
  const int pbase = blockIdx.x * 16;
  const int lane = tid & 31;
  const int w = tid >> 5;        // wave id == M-tile
  const int nloc = lane & 15;    // N within 16x16 tile
  const int h = lane >> 4;       // lane half

  // staging assignment: thread -> (row, column-half)
  const int r = tid >> 1;        // 0..127 : row = point*8 + k
  const int ch = tid & 1;
  const int p_r = r >> 3;
  const int gp_r = scene * NN + pbase + p_r;
  const int nb_r = knn[gp_r * KNB + (r & 7)];

  if (tid < 128) {  // relative coords for row tid
    int rr = tid;
    int gpp = scene * NN + pbase + (rr >> 3);
    int nbb = knn[gpp * KNB + (rr & 7)];
    relc[rr * 4 + 0] = coords[nbb * 4 + 1] - coords[gpp * 4 + 1];
    relc[rr * 4 + 1] = coords[nbb * 4 + 2] - coords[gpp * 4 + 2];
    relc[rr * 4 + 2] = coords[nbb * 4 + 3] - coords[gpp * 4 + 3];
  }

  // ---- stage A = nf (neighbor features), bf16 ----
  {
    const v4f* nf4 = (const v4f*)(F + nb_r * CC) + ch * 16;
    __bf16* dst = As + r * CC + ch * 64;
#pragma unroll
    for (int i = 0; i < 16; ++i) {
      v4f v = nf4[i];
      union { __bf16 hh[4]; unsigned long long u; } pk;
      pk.hh[0] = (__bf16)v.x; pk.hh[1] = (__bf16)v.y;
      pk.hh[2] = (__bf16)v.z; pk.hh[3] = (__bf16)v.w;
      *(unsigned long long*)(dst + 4 * i) = pk.u;
    }
  }
  __syncthreads();

  // ---- GEMM A: accF = nf @ W_ft ----
  v8f accF[8], accL[8];
#pragma unroll
  for (int j = 0; j < 8; ++j) {
    v8f z = {0.f, 0.f, 0.f, 0.f, 0.f, 0.f, 0.f, 0.f};
    accF[j] = z; accL[j] = z;
  }
  const __bf16* arow = As + (16 * w + nloc) * CC;
  gemm_tile(arow, WTft, nloc, h, accF);
  __syncthreads();

  // ---- restage A = rel_f = nf - f_center ----
  {
    const v4f* nf4 = (const v4f*)(F + nb_r * CC) + ch * 16;
    const v4f* cf4 = (const v4f*)(F + gp_r * CC) + ch * 16;
    __bf16* dst = As + r * CC + ch * 64;
#pragma unroll
    for (int i = 0; i < 16; ++i) {
      v4f v = nf4[i] - cf4[i];
      union { __bf16 hh[4]; unsigned long long u; } pk;
      pk.hh[0] = (__bf16)v.x; pk.hh[1] = (__bf16)v.y;
      pk.hh[2] = (__bf16)v.z; pk.hh[3] = (__bf16)v.w;
      *(unsigned long long*)(dst + 4 * i) = pk.u;
    }
  }
  __syncthreads();

  // ---- GEMM B: accL = rel_f @ W_feat ----
  gemm_tile(arow, WTfeat, nloc, h, accL);

  // ---- epilogue: per-lane softmax over K, weighted sum, residual, LN ----
  // D layout: acc[j][v] = S[row 16w + 8h + v, col 16j + nloc]
  //          -> point = 2w + h, neighbor k = v. Fully in-lane.
  const int p = 2 * w + h;
  const int gp = scene * NN + pbase + p;
  float rcx[8], rcy[8], rcz[8];
#pragma unroll
  for (int v = 0; v < 8; ++v) {
    rcx[v] = relc[(p * 8 + v) * 4 + 0];
    rcy[v] = relc[(p * 8 + v) * 4 + 1];
    rcz[v] = relc[(p * 8 + v) * 4 + 2];
  }
  float res[8];
  float sum = 0.f, sumsq = 0.f;
#pragma unroll
  for (int j = 0; j < 8; ++j) {
    int c = 16 * j + nloc;
    float wc0 = Wc[c], wc1 = Wc[128 + c], wc2 = Wc[256 + c];
    float bc = b_coord[c], bfe = b_feat[c], bft = b_ft[c];
    float lg[8];
    float mx = -3.0e38f;
#pragma unroll
    for (int v = 0; v < 8; ++v) {
      float lc = fmaf(rcz[v], wc2, fmaf(rcy[v], wc1, fmaf(rcx[v], wc0, bc)));
      float l = lc * (accL[j][v] + bfe) * 0.35355339059327373f;  // /sqrt(8)
      lg[v] = l;
      mx = fmaxf(mx, l);
    }
    float s = 0.f, e[8];
#pragma unroll
    for (int v = 0; v < 8; ++v) { e[v] = __expf(lg[v] - mx); s += e[v]; }
    float o = 0.f;
#pragma unroll
    for (int v = 0; v < 8; ++v) o = fmaf(e[v], accF[j][v] + bft, o);
    o /= s;
    float rr = o + F[gp * CC + c];  // residual
    res[j] = rr;
    sum += rr;
    sumsq = fmaf(rr, rr, sumsq);
  }
  // LayerNorm reduction: 16-lane half-wave owns the full 128-col row.
#pragma unroll
  for (int m = 1; m < 16; m <<= 1) {
    sum += __shfl_xor(sum, m, 32);
    sumsq += __shfl_xor(sumsq, m, 32);
  }
  float mean = sum * (1.0f / 128.0f);
  float var = sumsq * (1.0f / 128.0f) - mean * mean;
  float rstd = rsqrtf(var + 1e-5f);
#pragma unroll
  for (int j = 0; j < 8; ++j) {
    int c = 16 * j + nloc;
    out[gp * CC + c] = (res[j] - mean) * rstd * gamma_[c] + beta_[c];
  }
}

// ---------------------------------------------------------------------------
extern "C" void kernel_launch(void* const* d_in, const int* in_sizes, int n_in,
                              void* d_out, int out_size, void* d_ws, size_t ws_size,
                              hipStream_t stream) {
  const float* features = (const float*)d_in[0];
  const float* coords   = (const float*)d_in[1];
  const float* W_ft     = (const float*)d_in[2];
  const float* b_ft     = (const float*)d_in[3];
  const float* W_coord  = (const float*)d_in[4];
  const float* b_coord  = (const float*)d_in[5];
  const float* W_feat   = (const float*)d_in[6];
  const float* b_feat   = (const float*)d_in[7];
  const float* ln_gamma = (const float*)d_in[8];
  const float* ln_beta  = (const float*)d_in[9];
  float* out = (float*)d_out;

  int* knn = (int*)d_ws;
  __bf16* WTfeat = (__bf16*)((char*)d_ws + (size_t)BB * NN * KNB * sizeof(int));
  __bf16* WTft = WTfeat + CC * CC;

  prep_wt_kernel<<<dim3((CC * CC + 255) / 256), 256, 0, stream>>>(W_feat, W_ft, WTfeat, WTft);
  knn_kernel<<<dim3((NN + 255) / 256, BB), 256, 0, stream>>>(coords, knn);
  attn_kernel<<<dim3(NN / 16, BB), 256, 0, stream>>>(
      features, coords, knn, WTfeat, WTft, W_coord, b_coord, b_feat, b_ft,
      ln_gamma, ln_beta, out);
}
